// SimpleTransFusionHead_20177756356608
// MI455X (gfx1250) — compile-verified, matrix operand order checked by reading, not run on hardware
//
#include <hip/hip_runtime.h>
#include <hip/hip_bf16.h>
#include <math.h>

// ---------------------------------------------------------------------------
// Problem constants (TransFusion head)
// ---------------------------------------------------------------------------
#define BB    2
#define CIN_  512
#define HH    180
#define WW    180
#define HW_   (HH * WW)          // 32400
#define HID_  128
#define NC_   10
#define NH_   8
#define HD_   16                 // head dim = HID/NH
#define FFN_  256
#define NPROP 200
#define PRED_CH 20               // 2+1+3+2+2+10

#define LDP   40                 // padded LDS row stride (halves): 80B, 16B-aligned,
                                 // breaks 64-bank conflicts vs 64B stride

typedef __attribute__((ext_vector_type(16))) _Float16 v16h;
typedef __attribute__((ext_vector_type(8)))  float    v8f;

// ---------------------------------------------------------------------------
// WMMA fragment helpers (CDNA5 wave32 layouts, 16-bit operand 16x32)
// Both A (MxK) and B (KxN, stored transposed as NxK) tiles are kept in LDS
// K-contiguous per row so each lane's 16 halves are two aligned 16B runs ->
// 2x ds_load_b128 per fragment.
// ---------------------------------------------------------------------------
__device__ inline v8f zero8() {
  v8f z;
#pragma unroll
  for (int e = 0; e < 8; ++e) z[e] = 0.f;
  return z;
}

// Tile: 16 rows x 32 K, row-major in LDS with row stride `ld` (halves).
// Used for A directly and for B via the transposed (NxK) staging.
__device__ inline v16h load_tile_rk(const _Float16* base, int ld) {
  const int lane = threadIdx.x & 31;
  const int m = lane & 15;
  const int koff = (lane >> 4) * 8;
  v16h out;
#pragma unroll
  for (int v = 0; v < 8; ++v) {
    const int k = ((v < 4) ? (2 * v) : (8 + 2 * v)) + koff;
    out[2 * v]     = base[m * ld + k];
    out[2 * v + 1] = base[m * ld + k + 1];
  }
  return out;
}

// ---------------------------------------------------------------------------
// Implicit-GEMM 3x3 SAME conv (NCHW, 180x180), f16 WMMA, f32 accumulate.
// Block: 256 threads = 8 waves. Block tile: M = MT*16 rows x 128 pixels.
// Each wave owns a 16-pixel column strip and all MT row tiles.
// ---------------------------------------------------------------------------
template <int MT>
__global__ __launch_bounds__(256) void conv3x3_wmma(
    const float* __restrict__ in, const float* __restrict__ w,
    const float* __restrict__ bias, const float* __restrict__ bng,
    const float* __restrict__ bnb, float* __restrict__ out,
    int CIN, int OCreal, int relu) {
  constexpr int M = MT * 16;
  const int Kdim = CIN * 9;
  __shared__ _Float16 As[M * LDP];      // [M][32] padded
  __shared__ _Float16 Bs[128 * LDP];    // [pixel][32] padded (K-contiguous)
  const int b = blockIdx.y;
  const int n0 = blockIdx.x * 128;
  const int tid = threadIdx.x;
  const int wave = tid >> 5;

  v8f acc[MT];
#pragma unroll
  for (int mt = 0; mt < MT; ++mt) acc[mt] = zero8();

  for (int k0 = 0; k0 < Kdim; k0 += 32) {
    for (int i = tid; i < M * 32; i += 256) {
      const int m = i >> 5, k = i & 31;
      float v = 0.f;
      if (m < OCreal && (k0 + k) < Kdim) v = w[(size_t)m * Kdim + k0 + k];
      As[m * LDP + k] = (_Float16)v;
    }
    // im2col B tile: global reads coalesced over pixels, LDS stores K-contig.
    for (int i = tid; i < 32 * 128; i += 256) {
      const int k = i >> 7, j = i & 127;
      const int p = n0 + j;
      const int gk = k0 + k;
      float v = 0.f;
      if (p < HW_ && gk < Kdim) {
        const int ic = gk / 9, r = gk % 9;
        const int yy = p / WW + r / 3 - 1;
        const int xx = p % WW + r % 3 - 1;
        if (yy >= 0 && yy < HH && xx >= 0 && xx < WW)
          v = in[(((size_t)b * CIN + ic) * HH + yy) * WW + xx];
      }
      Bs[j * LDP + k] = (_Float16)v;
    }
    __syncthreads();
    const v16h bf = load_tile_rk(&Bs[(wave * 16) * LDP], LDP);
#pragma unroll
    for (int mt = 0; mt < MT; ++mt) {
      const v16h af = load_tile_rk(&As[(mt * 16) * LDP], LDP);
      acc[mt] = __builtin_amdgcn_wmma_f32_16x16x32_f16(
          false, af, false, bf, (short)0, acc[mt], false, false);
    }
    __syncthreads();
  }

  const int lane = tid & 31;
  const int p = n0 + wave * 16 + (lane & 15);
  const int rbase = (lane >> 4) * 8;
#pragma unroll
  for (int mt = 0; mt < MT; ++mt) {
#pragma unroll
    for (int i = 0; i < 8; ++i) {
      const int m = mt * 16 + rbase + i;
      if (m < OCreal && p < HW_) {
        float v = acc[mt][i] + (bias ? bias[m] : 0.f);
        if (bng) v = v * bng[m] + bnb[m];
        if (relu) v = fmaxf(v, 0.f);
        out[((size_t)b * OCreal + m) * HW_ + p] = v;
      }
    }
  }
}

// ---------------------------------------------------------------------------
// Batched GEMM via WMMA: out[b] = act(A @ Bsrc[b] + bias), A [M x K] f32,
// Bsrc [BB x K x N] f32, channel-major output [BB x M x N].
// ---------------------------------------------------------------------------
template <int MT>
__global__ __launch_bounds__(256) void gemm_wmma(
    const float* __restrict__ A, const float* __restrict__ Bsrc,
    const float* __restrict__ bias, float* __restrict__ out,
    int Kdim, int N, int OCreal, int relu) {
  constexpr int M = MT * 16;
  __shared__ _Float16 As[M * LDP];
  __shared__ _Float16 Bs[128 * LDP];
  const int b = blockIdx.y;
  const int n0 = blockIdx.x * 128;
  const int tid = threadIdx.x;
  const int wave = tid >> 5;

  v8f acc[MT];
#pragma unroll
  for (int mt = 0; mt < MT; ++mt) acc[mt] = zero8();

  for (int k0 = 0; k0 < Kdim; k0 += 32) {
    for (int i = tid; i < M * 32; i += 256) {
      const int m = i >> 5, k = i & 31;
      float v = 0.f;
      if (m < OCreal && (k0 + k) < Kdim) v = A[(size_t)m * Kdim + k0 + k];
      As[m * LDP + k] = (_Float16)v;
    }
    for (int i = tid; i < 32 * 128; i += 256) {
      const int k = i >> 7, j = i & 127;
      const int p = n0 + j;
      const int gk = k0 + k;
      float v = 0.f;
      if (p < N && gk < Kdim) v = Bsrc[((size_t)b * Kdim + gk) * N + p];
      Bs[j * LDP + k] = (_Float16)v;
    }
    __syncthreads();
    const v16h bf = load_tile_rk(&Bs[(wave * 16) * LDP], LDP);
#pragma unroll
    for (int mt = 0; mt < MT; ++mt) {
      const v16h af = load_tile_rk(&As[(mt * 16) * LDP], LDP);
      acc[mt] = __builtin_amdgcn_wmma_f32_16x16x32_f16(
          false, af, false, bf, (short)0, acc[mt], false, false);
    }
    __syncthreads();
  }

  const int lane = tid & 31;
  const int p = n0 + wave * 16 + (lane & 15);
  const int rbase = (lane >> 4) * 8;
#pragma unroll
  for (int mt = 0; mt < MT; ++mt) {
#pragma unroll
    for (int i = 0; i < 8; ++i) {
      const int m = mt * 16 + rbase + i;
      if (m < OCreal && p < N) {
        float v = acc[mt][i] + (bias ? bias[m] : 0.f);
        if (relu) v = fmaxf(v, 0.f);
        out[((size_t)b * OCreal + m) * N + p] = v;
      }
    }
  }
}

// ---------------------------------------------------------------------------
// sigmoid + 3x3 NMS mask (classes 8,9 use kernel 1; borders suppressed)
// ---------------------------------------------------------------------------
__device__ inline float sigm(float x) { return 1.f / (1.f + __expf(-x)); }

__global__ void k_nms(const float* __restrict__ dense, float* __restrict__ masked) {
  const int i = blockIdx.x * blockDim.x + threadIdx.x;
  if (i >= BB * NC_ * HW_) return;
  const int p = i % HW_;
  const int c = (i / HW_) % NC_;
  const int b = i / (HW_ * NC_);
  const int y = p / WW, x = p % WW;
  const float hm = sigm(dense[i]);
  float keep;
  if (c == 8 || c == 9) {
    keep = hm;
  } else if (y < 1 || y > HH - 2 || x < 1 || x > WW - 2) {
    keep = 0.f;  // reference leaves local_max==0 on the border
  } else {
    float mx = -1e30f;
    const float* base = dense + ((size_t)b * NC_ + c) * HW_;
#pragma unroll
    for (int dy = -1; dy <= 1; ++dy)
#pragma unroll
      for (int dx = -1; dx <= 1; ++dx)
        mx = fmaxf(mx, sigm(base[(y + dy) * WW + (x + dx)]));
    keep = (hm == mx) ? hm : 0.f;
  }
  masked[i] = keep;
}

// Iterative top-200 per batch; one 1024-thread block per batch (mutates masked).
__global__ __launch_bounds__(1024) void k_topk(float* __restrict__ masked,
                                               int* __restrict__ topidx) {
  __shared__ float sv[1024];
  __shared__ int   si[1024];
  const int b = blockIdx.x;
  const int t = threadIdx.x;
  float* base = masked + (size_t)b * (NC_ * HW_);
  for (int r = 0; r < NPROP; ++r) {
    float bv = -1e30f; int bi = 0;
    for (int i = t; i < NC_ * HW_; i += 1024) {
      const float v = base[i];
      if (v > bv) { bv = v; bi = i; }
    }
    sv[t] = bv; si[t] = bi;
    __syncthreads();
    for (int s = 512; s > 0; s >>= 1) {
      if (t < s && sv[t + s] > sv[t]) { sv[t] = sv[t + s]; si[t] = si[t + s]; }
      __syncthreads();
    }
    if (t == 0) { topidx[b * NPROP + r] = si[0]; base[si[0]] = -1e30f; }
    __threadfence_block();
    __syncthreads();
  }
}

__global__ void k_qpos(const int* __restrict__ topidx, float* __restrict__ qpos) {
  const int i = blockIdx.x * blockDim.x + threadIdx.x;
  if (i >= BB * NPROP) return;
  const int sp = topidx[i] % HW_;
  qpos[i * 2 + 0] = (float)(sp % WW) + 0.5f;
  qpos[i * 2 + 1] = (float)(sp / WW) + 0.5f;
}

// qf0[b,c,n] = lidar[b,c,idx] + ce_w[c,cls] + ce_b[c]
__global__ void k_gather(const float* __restrict__ lidar, const int* __restrict__ topidx,
                         const float* __restrict__ cew, const float* __restrict__ ceb,
                         float* __restrict__ qf0) {
  const int i = blockIdx.x * blockDim.x + threadIdx.x;
  if (i >= BB * HID_ * NPROP) return;
  const int n = i % NPROP;
  const int c = (i / NPROP) % HID_;
  const int b = i / (NPROP * HID_);
  const int top = topidx[b * NPROP + n];
  const int cls = top / HW_, sp = top % HW_;
  qf0[i] = lidar[((size_t)b * HID_ + c) * HW_ + sp] + cew[c * NC_ + cls] + ceb[c];
}

// posembed layer 1: relu(bn(conv1d(2->128,k1))). pos==nullptr -> BEV grid coords
__global__ void k_pe1(const float* __restrict__ pos, const float* __restrict__ w1,
                      const float* __restrict__ b1, const float* __restrict__ g,
                      const float* __restrict__ bet, float* __restrict__ out, int N) {
  const int i = blockIdx.x * blockDim.x + threadIdx.x;
  if (i >= BB * HID_ * N) return;
  const int n = i % N;
  const int h = (i / N) % HID_;
  const int b = i / (N * HID_);
  float px, py;
  if (pos) { px = pos[((size_t)b * N + n) * 2]; py = pos[((size_t)b * N + n) * 2 + 1]; }
  else     { px = (float)(n % WW) + 0.5f;      py = (float)(n / WW) + 0.5f; }
  float v = w1[h * 2] * px + w1[h * 2 + 1] * py + b1[h];
  v = v * g[h] + bet[h];
  out[((size_t)b * HID_ + h) * N + n] = fmaxf(v, 0.f);
}

// channel-major linear: out[b,o,n] = act(sum_c W[o,c] in[b,c,n] + bias[o]) (+resid)
__global__ void k_linear_cm(const float* __restrict__ in, const float* __restrict__ W,
                            const float* __restrict__ bias, const float* __restrict__ resid,
                            float* __restrict__ out, int Cin, int Cout, int N, int relu) {
  const int i = blockIdx.x * blockDim.x + threadIdx.x;
  if (i >= BB * Cout * N) return;
  const int n = i % N;
  const int o = (i / N) % Cout;
  const int b = i / (N * Cout);
  const float* x = in + (size_t)b * Cin * N + n;
  const float* w = W + (size_t)o * Cin;
  float s = bias ? bias[o] : 0.f;
  for (int c = 0; c < Cin; ++c) s += w[c] * x[(size_t)c * N];
  if (resid) s += resid[i];
  if (relu) s = fmaxf(s, 0.f);
  out[i] = s;
}

__global__ void k_add(const float* __restrict__ a, const float* __restrict__ b,
                      float* __restrict__ out, int len) {
  const int i = blockIdx.x * blockDim.x + threadIdx.x;
  if (i < len) out[i] = a[i] + b[i];
}

// LayerNorm over the channel axis of channel-major [BB,128,N]
__global__ void k_layernorm(const float* __restrict__ in, const float* __restrict__ g,
                            const float* __restrict__ bet, float* __restrict__ out, int N) {
  const int t = blockIdx.x * blockDim.x + threadIdx.x;
  if (t >= BB * N) return;
  const int b = t / N, n = t % N;
  const float* x = in + (size_t)b * HID_ * N + n;
  float mean = 0.f;
  for (int c = 0; c < HID_; ++c) mean += x[(size_t)c * N];
  mean *= (1.f / HID_);
  float var = 0.f;
  for (int c = 0; c < HID_; ++c) { const float d = x[(size_t)c * N] - mean; var += d * d; }
  var *= (1.f / HID_);
  const float inv = rsqrtf(var + 1e-5f);
  for (int c = 0; c < HID_; ++c)
    out[((size_t)b * HID_ + c) * N + n] = (x[(size_t)c * N] - mean) * inv * g[c] + bet[c];
}

// One wave per (b, head, query); online-softmax streaming over Nk keys.
// Q,K,V,O all channel-major [BB, 128, *].
__global__ __launch_bounds__(256) void k_attn(
    const float* __restrict__ Qc, const float* __restrict__ Kc,
    const float* __restrict__ Vc, float* __restrict__ Oc, int Nk, int Nq) {
  const int wid = (blockIdx.x * blockDim.x + threadIdx.x) >> 5;
  const int lane = threadIdx.x & 31;
  if (wid >= BB * NH_ * Nq) return;
  const int qi = wid % Nq;
  const int h = (wid / Nq) % NH_;
  const int b = wid / (Nq * NH_);
  const float scale = 0.25f;  // 1/sqrt(16)
  float qv[HD_];
#pragma unroll
  for (int d = 0; d < HD_; ++d)
    qv[d] = Qc[((size_t)b * HID_ + h * HD_ + d) * Nq + qi];
  float m = -1e30f, l = 0.f, acc[HD_];
#pragma unroll
  for (int d = 0; d < HD_; ++d) acc[d] = 0.f;
  for (int n = lane; n < Nk; n += 32) {
    if (n + 32 < Nk) {  // lowers to global_prefetch_b8
      __builtin_prefetch(&Kc[((size_t)b * HID_ + h * HD_) * Nk + n + 32], 0, 0);
      __builtin_prefetch(&Vc[((size_t)b * HID_ + h * HD_) * Nk + n + 32], 0, 0);
    }
    float s = 0.f;
#pragma unroll
    for (int d = 0; d < HD_; ++d)
      s += qv[d] * Kc[((size_t)b * HID_ + h * HD_ + d) * Nk + n];
    s *= scale;
    const float nm = fmaxf(m, s);
    const float corr = __expf(m - nm);
    const float p = __expf(s - nm);
    l = l * corr + p;
#pragma unroll
    for (int d = 0; d < HD_; ++d)
      acc[d] = acc[d] * corr + p * Vc[((size_t)b * HID_ + h * HD_ + d) * Nk + n];
    m = nm;
  }
#pragma unroll
  for (int off = 16; off >= 1; off >>= 1) {
    const float m2 = __shfl_xor(m, off, 32);
    const float l2 = __shfl_xor(l, off, 32);
    const float nm = fmaxf(m, m2);
    const float c1 = __expf(m - nm), c2 = __expf(m2 - nm);
    l = l * c1 + l2 * c2;
#pragma unroll
    for (int d = 0; d < HD_; ++d) {
      const float a2 = __shfl_xor(acc[d], off, 32);
      acc[d] = acc[d] * c1 + a2 * c2;
    }
    m = nm;
  }
  if (lane == 0) {
    const float inv = 1.f / l;
#pragma unroll
    for (int d = 0; d < HD_; ++d)
      Oc[((size_t)b * HID_ + h * HD_ + d) * Nq + qi] = acc[d] * inv;
  }
}

// conv1d kernel-3 SAME over 200 tokens (channel-major), optional BN/ReLU,
// writes into out at channel offset ch_off of a ch_tot-channel tensor;
// addpos (qpos) added for the center head.
__global__ void k_conv1d3(const float* __restrict__ in, const float* __restrict__ w,
                          const float* __restrict__ bias, const float* __restrict__ g,
                          const float* __restrict__ bet, float* __restrict__ out,
                          int Cout, int relu, int ch_off, int ch_tot,
                          const float* __restrict__ addpos) {
  const int i = blockIdx.x * blockDim.x + threadIdx.x;
  if (i >= BB * Cout * NPROP) return;
  const int n = i % NPROP;
  const int o = (i / NPROP) % Cout;
  const int b = i / (NPROP * Cout);
  float s = bias[o];
  for (int c = 0; c < HID_; ++c) {
    const float* wr = w + ((size_t)o * HID_ + c) * 3;
    const float* xr = in + ((size_t)b * HID_ + c) * NPROP;
    if (n > 0) s += wr[0] * xr[n - 1];
    s += wr[1] * xr[n];
    if (n < NPROP - 1) s += wr[2] * xr[n + 1];
  }
  if (g) s = s * g[o] + bet[o];
  if (relu) s = fmaxf(s, 0.f);
  if (addpos) s += addpos[((size_t)b * NPROP + n) * 2 + o];
  out[((size_t)b * ch_tot + ch_off + o) * NPROP + n] = s;
}

// ---------------------------------------------------------------------------
// Orchestration
// ---------------------------------------------------------------------------
static inline dim3 gr1(long n, int blk = 256) { return dim3((unsigned)((n + blk - 1) / blk)); }

extern "C" void kernel_launch(void* const* d_in, const int* in_sizes, int n_in,
                              void* d_out, int out_size, void* d_ws, size_t ws_size,
                              hipStream_t stream) {
  (void)in_sizes; (void)n_in; (void)out_size; (void)ws_size;
  const float* F[128];
  for (int i = 0; i < n_in && i < 128; ++i) F[i] = (const float*)d_in[i];

  // parameter indices (setup_inputs dict order)
  const float* inputs   = F[0];
  const float* shared_w = F[1];  const float* shared_b = F[2];
  const float* hm1_w = F[3];  const float* hm1_b = F[4];
  const float* hm1_g = F[5];  const float* hm1_be = F[6];
  const float* hm2_w = F[7];  const float* hm2_b = F[8];
  const float* ce_w = F[9];   const float* ce_b = F[10];
  const float* pes_w1 = F[11]; const float* pes_b1 = F[12];
  const float* pes_g = F[13];  const float* pes_be = F[14];
  const float* pes_w2 = F[15]; const float* pes_b2 = F[16];
  const float* pec_w1 = F[17]; const float* pec_b1 = F[18];
  const float* pec_g = F[19];  const float* pec_be = F[20];
  const float* pec_w2 = F[21]; const float* pec_b2 = F[22];
  const float* sa_wi = F[23]; const float* sa_bi = F[24];
  const float* sa_wo = F[25]; const float* sa_bo = F[26];
  const float* ca_wi = F[27]; const float* ca_bi = F[28];
  const float* ca_wo = F[29]; const float* ca_bo = F[30];
  const float* ff_w1 = F[31]; const float* ff_b1 = F[32];
  const float* ff_w2 = F[33]; const float* ff_b2 = F[34];
  const float* n1_g = F[35]; const float* n1_b = F[36];
  const float* n2_g = F[37]; const float* n2_b = F[38];
  const float* n3_g = F[39]; const float* n3_b = F[40];
  // heads: base 41 + hidx*6 -> w1,b1,g,beta,w2,b2

  float* preds = (float*)d_out;                         // [2,20,200]
  float* dense = preds + (size_t)BB * PRED_CH * NPROP;  // [2,10,180,180]

  // workspace carving
  float* wsf = (float*)d_ws;
  size_t off = 0;
  const size_t BIG = (size_t)BB * HID_ * HW_;  // 8.29M floats
  const size_t SML = (size_t)BB * HID_ * NPROP;
  float* lidar = wsf + off; off += BIG;   // shared conv out (kept)
  float* bufA  = wsf + off; off += BIG;   // h -> x1(bev) -> Kproj
  float* bufB  = wsf + off; off += BIG;   // kpe -> kk (in-place)
  float* bufC  = wsf + off; off += BIG;   // Vproj
  float* masked = wsf + off; off += (size_t)BB * NC_ * HW_;
  float* qpos = wsf + off; off += (size_t)BB * NPROP * 2;
  float* x1q  = wsf + off; off += SML;
  float* qf0  = wsf + off; off += SML;
  float* qpe  = wsf + off; off += SML;
  float* tbuf = wsf + off; off += SML;
  float* Qc   = wsf + off; off += SML;
  float* Kcs  = wsf + off; off += SML;
  float* Vcs  = wsf + off; off += SML;
  float* Ocb  = wsf + off; off += SML;
  float* q1   = wsf + off; off += SML;
  float* q1ln = wsf + off; off += SML;
  float* q2ln = wsf + off; off += SML;
  float* f1   = wsf + off; off += (size_t)BB * FFN_ * NPROP;
  float* qfin = wsf + off; off += SML;
  float* headx = wsf + off; off += SML;
  int* topidx = (int*)(wsf + off); off += BB * NPROP;

  const dim3 gconv((HW_ + 127) / 128, BB);

  // 1) shared conv 512->128 (bias only)
  conv3x3_wmma<8><<<gconv, 256, 0, stream>>>(inputs, shared_w, shared_b,
                                             nullptr, nullptr, lidar, CIN_, HID_, 0);
  // 2) hm1 conv 128->128 + BN + ReLU
  conv3x3_wmma<8><<<gconv, 256, 0, stream>>>(lidar, hm1_w, hm1_b,
                                             hm1_g, hm1_be, bufA, HID_, HID_, 1);
  // 3) hm2 conv 128->10 -> dense (straight into d_out)
  conv3x3_wmma<1><<<gconv, 256, 0, stream>>>(bufA, hm2_w, hm2_b,
                                             nullptr, nullptr, dense, HID_, NC_, 0);
  // 4) sigmoid + NMS mask
  k_nms<<<gr1((long)BB * NC_ * HW_), 256, 0, stream>>>(dense, masked);
  // 5) top-200 per batch
  k_topk<<<BB, 1024, 0, stream>>>(masked, topidx);
  // 6) query positions + feature gather (+class embedding)
  k_qpos<<<gr1(BB * NPROP), 256, 0, stream>>>(topidx, qpos);
  k_gather<<<gr1((long)BB * HID_ * NPROP), 256, 0, stream>>>(lidar, topidx, ce_w, ce_b, qf0);

  // 7) query pos-embed (200 tokens: plain kernels)
  k_pe1<<<gr1((long)BB * HID_ * NPROP), 256, 0, stream>>>(qpos, pes_w1, pes_b1, pes_g, pes_be, x1q, NPROP);
  k_linear_cm<<<gr1((long)BB * HID_ * NPROP), 256, 0, stream>>>(x1q, pes_w2, pes_b2, nullptr, qpe,
                                                               HID_, HID_, NPROP, 0);
  // 8) BEV key pos-embed: layer1 elementwise, layer2 WMMA GEMM (128x128x32400)
  k_pe1<<<gr1((long)BB * HID_ * HW_), 256, 0, stream>>>(nullptr, pec_w1, pec_b1, pec_g, pec_be, bufA, HW_);
  gemm_wmma<8><<<gconv, 256, 0, stream>>>(pec_w2, bufA, pec_b2, bufB, HID_, HW_, HID_, 0);

  // 9) self-attention over 200 queries
  k_add<<<gr1((long)SML), 256, 0, stream>>>(qf0, qpe, tbuf, (int)SML);
  k_linear_cm<<<gr1((long)SML), 256, 0, stream>>>(tbuf, sa_wi, sa_bi, nullptr, Qc, HID_, HID_, NPROP, 0);
  k_linear_cm<<<gr1((long)SML), 256, 0, stream>>>(tbuf, sa_wi + HID_ * HID_, sa_bi + HID_, nullptr, Kcs, HID_, HID_, NPROP, 0);
  k_linear_cm<<<gr1((long)SML), 256, 0, stream>>>(tbuf, sa_wi + 2 * HID_ * HID_, sa_bi + 2 * HID_, nullptr, Vcs, HID_, HID_, NPROP, 0);
  k_attn<<<gr1((long)BB * NH_ * NPROP * 32), 256, 0, stream>>>(Qc, Kcs, Vcs, Ocb, NPROP, NPROP);
  k_linear_cm<<<gr1((long)SML), 256, 0, stream>>>(Ocb, sa_wo, sa_bo, qf0, q1, HID_, HID_, NPROP, 0);
  k_layernorm<<<gr1(BB * NPROP), 256, 0, stream>>>(q1, n1_g, n1_b, q1ln, NPROP);

  // 10) cross-attention over 32400 BEV keys
  k_add<<<gr1((long)BIG), 256, 0, stream>>>(bufB, lidar, bufB, (int)BIG);       // kk = key + kpe
  k_add<<<gr1((long)SML), 256, 0, stream>>>(q1ln, qpe, tbuf, (int)SML);         // q + qpe
  k_linear_cm<<<gr1((long)SML), 256, 0, stream>>>(tbuf, ca_wi, ca_bi, nullptr, Qc, HID_, HID_, NPROP, 0);
  gemm_wmma<8><<<gconv, 256, 0, stream>>>(ca_wi + HID_ * HID_, bufB, ca_bi + HID_, bufA, HID_, HW_, HID_, 0);          // Kproj
  gemm_wmma<8><<<gconv, 256, 0, stream>>>(ca_wi + 2 * HID_ * HID_, bufB, ca_bi + 2 * HID_, bufC, HID_, HW_, HID_, 0);  // Vproj
  k_attn<<<gr1((long)BB * NH_ * NPROP * 32), 256, 0, stream>>>(Qc, bufA, bufC, Ocb, HW_, NPROP);
  k_linear_cm<<<gr1((long)SML), 256, 0, stream>>>(Ocb, ca_wo, ca_bo, q1ln, q1, HID_, HID_, NPROP, 0);
  k_layernorm<<<gr1(BB * NPROP), 256, 0, stream>>>(q1, n2_g, n2_b, q2ln, NPROP);

  // 11) FFN + LN3
  k_linear_cm<<<gr1((long)BB * FFN_ * NPROP), 256, 0, stream>>>(q2ln, ff_w1, ff_b1, nullptr, f1, HID_, FFN_, NPROP, 1);
  k_linear_cm<<<gr1((long)SML), 256, 0, stream>>>(f1, ff_w2, ff_b2, q2ln, q1, FFN_, HID_, NPROP, 0);
  k_layernorm<<<gr1(BB * NPROP), 256, 0, stream>>>(q1, n3_g, n3_b, qfin, NPROP);

  // 12) prediction heads (conv1d k=3), center gets +qpos
  const int head_cout[6] = {2, 1, 3, 2, 2, NC_};
  const int head_off[6]  = {0, 2, 3, 6, 8, 10};
  for (int hx = 0; hx < 6; ++hx) {
    const float* w1 = F[41 + hx * 6 + 0];
    const float* b1 = F[41 + hx * 6 + 1];
    const float* g  = F[41 + hx * 6 + 2];
    const float* be = F[41 + hx * 6 + 3];
    const float* w2 = F[41 + hx * 6 + 4];
    const float* b2 = F[41 + hx * 6 + 5];
    k_conv1d3<<<gr1((long)BB * HID_ * NPROP), 256, 0, stream>>>(
        qfin, w1, b1, g, be, headx, HID_, 1, 0, HID_, nullptr);
    k_conv1d3<<<gr1((long)BB * head_cout[hx] * NPROP), 256, 0, stream>>>(
        headx, w2, b2, nullptr, nullptr, preds, head_cout[hx], 0,
        head_off[hx], PRED_CH, (hx == 0) ? qpos : nullptr);
  }
}
